// TransformerLayer_47553877901673
// MI455X (gfx1250) — compile-verified, matrix operand order checked by reading, not run on hardware
//
#include <hip/hip_runtime.h>
#include <hip/hip_bf16.h>
#include <math.h>
#include <stdint.h>

// ---------------------------------------------------------------------------
// Transformer layer for MI455X (gfx1250, wave32, WMMA 16x16x32 f16->f32,
// async global->LDS double-buffered GEMM pipeline).
// ---------------------------------------------------------------------------

typedef __attribute__((ext_vector_type(16))) _Float16 v16h;
typedef __attribute__((ext_vector_type(8)))  _Float16 h8v;
typedef __attribute__((ext_vector_type(8)))  float    v8f;

#define DMODEL 512
#define NHEADS 8
#define DK     64
#define DFF    2048
#define BATCH  8
#define SEQ    1024
#define ROWS   (BATCH * SEQ)     // 8192
#define NEGVAL (-1e32f)

static __device__ __forceinline__ v8f wmma_f16(v16h a, v16h b, v8f c) {
  // D = A(16x32 f16) * B(32x16 f16) + C(16x16 f32)
  return __builtin_amdgcn_wmma_f32_16x16x32_f16(false, a, false, b, (short)0, c,
                                                false, false);
}

// Load a 16x32 f16 A-fragment (row-major source, leading dim ld).
// ISA 7.12.2: lanes 0-15 hold rows 0-15 / K 0-7 & 16-23; lanes 16-31 hold
// rows 0-15 / K 8-15 & 24-31.  Same pattern serves as a B-fragment when the
// source is stored [n][k] (B transposed), since B column n <-> "row" n.
static __device__ __forceinline__ v16h load_frag(const _Float16* base, int ld,
                                                 int row, int k0, int lane) {
  const _Float16* p = base + (size_t)(row + (lane & 15)) * (size_t)ld
                           + (size_t)(k0 + ((lane >> 4) << 3));
  h8v lo = *(const h8v*)p;         // k0 + kb .. +7
  h8v hi = *(const h8v*)(p + 16);  // k0 + kb + 16 .. +23
  v16h r;
#pragma unroll
  for (int i = 0; i < 8; ++i) { r[i] = lo[i]; r[8 + i] = hi[i]; }
  return r;
}

// Async copy of one 16-byte chunk global -> LDS (CDNA5 ASYNCcnt path).
static __device__ __forceinline__ void async_cp16(const _Float16* gsrc,
                                                  _Float16* lds_dst) {
  unsigned lds_off = (unsigned)(uintptr_t)lds_dst;      // addr[31:0] = LDS off
  unsigned long long ga = (unsigned long long)(uintptr_t)gsrc;
  asm volatile("global_load_async_to_lds_b128 %0, %1, off"
               :: "v"(lds_off), "v"(ga) : "memory");
}
static __device__ __forceinline__ void wait_async0() {
  asm volatile("s_wait_asynccnt 0x0" ::: "memory");
}

// Row reductions across the 16 lanes of a half-wave (C-layout: each row's 16
// columns live in lanes 0-15 or 16-31; xor masks <=8 never cross halves).
static __device__ __forceinline__ float redmax16(float v) {
#pragma unroll
  for (int m = 1; m < 16; m <<= 1) v = fmaxf(v, __shfl_xor(v, m, 32));
  return v;
}
static __device__ __forceinline__ float redsum16(float v) {
#pragma unroll
  for (int m = 1; m < 16; m <<= 1) v += __shfl_xor(v, m, 32);
  return v;
}

// ---------------------------------------------------------------------------
// Prep kernels
// ---------------------------------------------------------------------------
__global__ void cvt_f32_to_f16(const float* __restrict__ s,
                               _Float16* __restrict__ d, int n) {
  int i = blockIdx.x * blockDim.x + threadIdx.x;
  if (i < n) d[i] = (_Float16)s[i];
}

// Wt[n][k] = (f16) W[k][n]   (W is K x N row-major)
__global__ void wtrans_kernel(const float* __restrict__ W,
                              _Float16* __restrict__ Wt, int K, int N) {
  int k = blockIdx.y * 16 + threadIdx.x;
  int n = blockIdx.x * 16 + threadIdx.y;
  if (k < K && n < N) Wt[(size_t)n * K + k] = (_Float16)W[(size_t)k * N + n];
}

// ---------------------------------------------------------------------------
// Pipelined block GEMM: C(64x64) per block of 4 waves.
//   A: M x K f16 row-major;  Bt: N x K f16 ("B transposed").
// Double-buffered async global->LDS staging of both 64x32 tiles, WMMA from
// LDS.  Epilogue modes: 0 = plain [M,N] f32/f16 (+bias, opt relu)
//                       1 = f16 [b,h,s,d] head-split store (q/k)
//                       2 = f16 [b,h,d,s] transposed store (v)
// grid = (N/64, M/64), block = 128.
// ---------------------------------------------------------------------------
__global__ void gemm_wmma(const _Float16* __restrict__ A,
                          const _Float16* __restrict__ Bt,
                          const float* __restrict__ bias,
                          float* __restrict__ outF,
                          _Float16* __restrict__ outH,
                          int N, int K, int relu, int mode) {
  __shared__ __align__(16) _Float16 sA[2][64 * 32];
  __shared__ __align__(16) _Float16 sB[2][64 * 32];

  const int tid = threadIdx.x;           // 0..127
  const int lane = tid & 31, wv = tid >> 5;
  const int m0 = blockIdx.y * 64, n0 = blockIdx.x * 64;
  const int nk = K / 32;

  // Each thread stages 2 x 16B chunks of the A tile and 2 of the B tile.
  // Tile = 64 rows x 32 halfs = 256 chunks of 16B.
  auto issue = [&](int buf, int k0) {
#pragma unroll
    for (int rep = 0; rep < 2; ++rep) {
      int c = tid + rep * 128;
      int row = c >> 2, off = (c & 3) * 8;   // off in halfs
      async_cp16(A + (size_t)(m0 + row) * K + k0 + off,
                 &sA[buf][row * 32 + off]);
      async_cp16(Bt + (size_t)(n0 + row) * K + k0 + off,
                 &sB[buf][row * 32 + off]);
    }
  };

  issue(0, 0);
  v8f c[4] = {};
  for (int ks = 0; ks < nk; ++ks) {
    wait_async0();          // this wave's async copies done
    __syncthreads();        // => whole tile resident
    if (ks + 1 < nk) issue((ks + 1) & 1, (ks + 1) * 32);
    const _Float16* pa = sA[ks & 1];
    const _Float16* pb = sB[ks & 1];
    // hoist all fragment loads -> one ds clause, staggered dscnt waits
    v16h a  = load_frag(pa, 32, wv * 16, 0, lane);
    v16h b0 = load_frag(pb, 32, 0,  0, lane);
    v16h b1 = load_frag(pb, 32, 16, 0, lane);
    v16h b2 = load_frag(pb, 32, 32, 0, lane);
    v16h b3 = load_frag(pb, 32, 48, 0, lane);
    c[0] = wmma_f16(a, b0, c[0]);
    c[1] = wmma_f16(a, b1, c[1]);
    c[2] = wmma_f16(a, b2, c[2]);
    c[3] = wmma_f16(a, b3, c[3]);
    __syncthreads();        // all waves done reading before buffer reuse
  }

  const int hl = lane >> 4, l15 = lane & 15;
  const int mbase = m0 + wv * 16;
#pragma unroll
  for (int t = 0; t < 4; ++t)
#pragma unroll
    for (int r = 0; r < 8; ++r) {
      int mrow = mbase + r + 8 * hl;
      int n    = n0 + t * 16 + l15;
      float v  = c[t][r] + bias[n];
      if (relu) v = fmaxf(v, 0.0f);
      if (mode == 0) {
        if (outF) outF[(size_t)mrow * N + n] = v;
        if (outH) outH[(size_t)mrow * N + n] = (_Float16)v;
      } else {
        int bb = mrow >> 10, s = mrow & 1023, hd = n >> 6, d = n & 63;
        if (mode == 1)
          outH[(((size_t)bb * NHEADS + hd) * SEQ + s) * DK + d] = (_Float16)v;
        else
          outH[(((size_t)bb * NHEADS + hd) * DK + d) * SEQ + s] = (_Float16)v;
      }
    }
}

// ---------------------------------------------------------------------------
// Flash-style attention.  One wave per (bh, 16-row query tile).
// grid = (SEQ/16, B*H), block = 32.
// ---------------------------------------------------------------------------
__global__ void attn_kernel(const _Float16* __restrict__ q16,
                            const _Float16* __restrict__ k16,
                            const _Float16* __restrict__ vT,
                            const float* __restrict__ dif,
                            const float* __restrict__ a_p,
                            const float* __restrict__ alpha_p,
                            const int* __restrict__ mask_p,
                            _Float16* __restrict__ ctx) {
  __shared__ __align__(16) _Float16 sP[16 * 32];   // staged P tile

  const int lane = threadIdx.x & 31;
  const int hl = lane >> 4, l15 = lane & 15;
  const int bh = blockIdx.y;
  const int bb = bh >> 3, hd = bh & 7;
  const int q0 = blockIdx.x * 16;
  const int maskv = *mask_p;

  const float a_sig = 1.0f / (1.0f + expf(-a_p[hd]));
  const float asp   = log1pf(expf(alpha_p[hd]));          // softplus
  const _Float16* qbase = q16 + (size_t)bh * SEQ * DK;
  const _Float16* kbase = k16 + (size_t)bh * SEQ * DK;
  const _Float16* vbase = vT  + (size_t)bh * DK * SEQ;

  v16h aq0 = load_frag(qbase, DK, q0, 0, lane);
  v16h aq1 = load_frag(qbase, DK, q0, 32, lane);

  v8f o[4] = {};
  float mrow[8], lrow[8];
#pragma unroll
  for (int r = 0; r < 8; ++r) { mrow[r] = -INFINITY; lrow[r] = 0.0f; }

  for (int kb = 0; kb < SEQ; kb += 32) {
    // ---- scores: two 16x16 tiles covering keys [kb, kb+32) ----
    v16h bk00 = load_frag(kbase, DK, kb,      0,  lane);
    v16h bk01 = load_frag(kbase, DK, kb,      32, lane);
    v16h bk10 = load_frag(kbase, DK, kb + 16, 0,  lane);
    v16h bk11 = load_frag(kbase, DK, kb + 16, 32, lane);
    v8f s0 = {}, s1 = {};
    s0 = wmma_f16(aq0, bk00, s0);
    s0 = wmma_f16(aq1, bk01, s0);
    s1 = wmma_f16(aq0, bk10, s1);
    s1 = wmma_f16(aq1, bk11, s1);

    float p0[8], p1[8];
#pragma unroll
    for (int r = 0; r < 8; ++r) {
      int i  = q0 + r + 8 * hl;
      int j0 = kb + l15;
      int j1 = j0 + 16;
      float dm0 = dif[((size_t)bb * SEQ + i) * SEQ + j0];
      float dm1 = dif[((size_t)bb * SEQ + i) * SEQ + j1];
      float dc0 = a_sig + (1.0f - a_sig) * expf(-asp * fabsf((float)(i - j0)));
      float dc1 = a_sig + (1.0f - a_sig) * expf(-asp * fabsf((float)(i - j1)));
      float v0 = s0[r] * 0.125f * dm0 * dc0;   // 1/sqrt(64)
      float v1 = s1[r] * 0.125f * dm1 * dc1;
      if (j0 > i + maskv - 1) v0 = NEGVAL;
      if (j1 > i + maskv - 1) v1 = NEGVAL;
      // online softmax update for row i
      float mv   = redmax16(fmaxf(v0, v1));
      float mnew = fmaxf(mrow[r], mv);
      float corr = expf(mrow[r] - mnew);       // 0 when mrow == -inf
      float e0 = expf(v0 - mnew);
      float e1 = expf(v1 - mnew);
      lrow[r] = lrow[r] * corr + redsum16(e0 + e1);
      mrow[r] = mnew;
#pragma unroll
      for (int t = 0; t < 4; ++t) o[t][r] *= corr;
      p0[r] = e0; p1[r] = e1;
    }

    // ---- reshape P (C-layout) -> A-layout through LDS ----
#pragma unroll
    for (int r = 0; r < 8; ++r) {
      int row = r + 8 * hl;
      sP[row * 32 + l15]      = (_Float16)p0[r];
      sP[row * 32 + 16 + l15] = (_Float16)p1[r];
    }
    __syncthreads();
    v16h ap = load_frag(sP, 32, 0, 0, lane);
    __syncthreads();

    // ---- O += P(16x32) x V(32x64) ----
    v16h bv0 = load_frag(vbase, SEQ, 0,  kb, lane);
    v16h bv1 = load_frag(vbase, SEQ, 16, kb, lane);
    v16h bv2 = load_frag(vbase, SEQ, 32, kb, lane);
    v16h bv3 = load_frag(vbase, SEQ, 48, kb, lane);
    o[0] = wmma_f16(ap, bv0, o[0]);
    o[1] = wmma_f16(ap, bv1, o[1]);
    o[2] = wmma_f16(ap, bv2, o[2]);
    o[3] = wmma_f16(ap, bv3, o[3]);
  }

  // ---- normalize + (mask==0 -> zero row 0) + store to concat buffer ----
#pragma unroll
  for (int t = 0; t < 4; ++t)
#pragma unroll
    for (int r = 0; r < 8; ++r) {
      int i = q0 + r + 8 * hl;
      float val = o[t][r] / lrow[r];
      if (maskv == 0 && i == 0) val = 0.0f;
      ctx[((size_t)bb * SEQ + i) * DMODEL + hd * DK + t * 16 + l15] =
          (_Float16)val;
    }
}

// ---------------------------------------------------------------------------
// Fused residual + LayerNorm over D=512.  One block (256 thr) per row.
// ---------------------------------------------------------------------------
__global__ void ln_kernel(const float* __restrict__ xa,
                          const float* __restrict__ xb,
                          const float* __restrict__ g,
                          const float* __restrict__ bet,
                          float* __restrict__ outF,
                          _Float16* __restrict__ outH) {
  __shared__ float ss[8], sq[8];
  const int row = blockIdx.x, tid = threadIdx.x;
  const size_t base = (size_t)row * DMODEL;
  float v0 = xa[base + tid]       + xb[base + tid];
  float v1 = xa[base + 256 + tid] + xb[base + 256 + tid];
  float s  = v0 + v1;
  float q  = v0 * v0 + v1 * v1;
#pragma unroll
  for (int m = 1; m < 32; m <<= 1) {
    s += __shfl_xor(s, m, 32);
    q += __shfl_xor(q, m, 32);
  }
  const int w = tid >> 5, ln = tid & 31;
  if (ln == 0) { ss[w] = s; sq[w] = q; }
  __syncthreads();
  if (tid == 0) {
    float S = 0.0f, Q = 0.0f;
#pragma unroll
    for (int i = 0; i < 8; ++i) { S += ss[i]; Q += sq[i]; }
    float mean = S * (1.0f / DMODEL);
    float var  = Q * (1.0f / DMODEL) - mean * mean;
    ss[0] = mean;
    sq[0] = rsqrtf(var + 1e-5f);
  }
  __syncthreads();
  const float mean = ss[0], rs = sq[0];
  float o0 = (v0 - mean) * rs * g[tid]       + bet[tid];
  float o1 = (v1 - mean) * rs * g[256 + tid] + bet[256 + tid];
  outF[base + tid]       = o0;
  outF[base + 256 + tid] = o1;
  if (outH) {
    outH[base + tid]       = (_Float16)o0;
    outH[base + 256 + tid] = (_Float16)o1;
  }
}

// ---------------------------------------------------------------------------
// Host side
// ---------------------------------------------------------------------------
extern "C" void kernel_launch(void* const* d_in, const int* in_sizes, int n_in,
                              void* d_out, int out_size, void* d_ws,
                              size_t ws_size, hipStream_t stream) {
  (void)in_sizes; (void)n_in; (void)out_size; (void)ws_size;
  const float* query  = (const float*)d_in[0];
  const float* key    = (const float*)d_in[1];
  const float* values = (const float*)d_in[2];
  const float* dif    = (const float*)d_in[3];
  const float* Wk     = (const float*)d_in[4];
  const float* bk     = (const float*)d_in[5];
  const float* Wv     = (const float*)d_in[6];
  const float* bv     = (const float*)d_in[7];
  const float* Wo     = (const float*)d_in[8];
  const float* bo     = (const float*)d_in[9];
  const float* a_p    = (const float*)d_in[10];
  const float* al_p   = (const float*)d_in[11];
  const float* ln1g   = (const float*)d_in[12];
  const float* ln1b   = (const float*)d_in[13];
  const float* W1     = (const float*)d_in[14];
  const float* b1     = (const float*)d_in[15];
  const float* W2     = (const float*)d_in[16];
  const float* b2     = (const float*)d_in[17];
  const float* ln2g   = (const float*)d_in[18];
  const float* ln2b   = (const float*)d_in[19];
  const int*   maskp  = (const int*)d_in[20];

  // workspace layout (bytes)
  char* p = (char*)d_ws;
  size_t off = 0;
  auto alloc = [&](size_t bytes) { char* r = p + off; off += bytes; return r; };
  const size_t SZ_W   = (size_t)DMODEL * DMODEL * 2;   // 512x512 f16
  const size_t SZ_WF  = (size_t)DMODEL * DFF * 2;      // 512x2048 f16
  const size_t SZ_ACT = (size_t)ROWS * DMODEL * 2;     // 8192x512 f16
  const size_t SZ_F32 = (size_t)ROWS * DMODEL * 4;     // 8192x512 f32

  _Float16* WkT   = (_Float16*)alloc(SZ_W);
  _Float16* WvT   = (_Float16*)alloc(SZ_W);
  _Float16* WoT   = (_Float16*)alloc(SZ_W);
  _Float16* W1T   = (_Float16*)alloc(SZ_WF);
  _Float16* W2T   = (_Float16*)alloc(SZ_WF);
  _Float16* qry16 = (_Float16*)alloc(SZ_ACT);
  _Float16* key16 = (_Float16*)alloc(SZ_ACT);
  _Float16* val16 = (_Float16*)alloc(SZ_ACT);
  _Float16* q16   = (_Float16*)alloc(SZ_ACT);   // [b,h,s,d]
  _Float16* k16   = (_Float16*)alloc(SZ_ACT);   // [b,h,s,d]
  _Float16* vT16  = (_Float16*)alloc(SZ_ACT);   // [b,h,d,s]
  _Float16* ctx16 = (_Float16*)alloc(SZ_ACT);   // concat heads
  float*    attnO = (float*)alloc(SZ_F32);      // attn out / later ffn out
  float*    xbuf  = (float*)alloc(SZ_F32);      // LN1 output f32
  _Float16* x16   = (_Float16*)alloc(SZ_ACT);   // LN1 output f16
  // h16 (8192x2048 f16 = 32MB) aliases dead qry16..q16 (exactly 32MB)
  _Float16* h16   = qry16;
  float*    ffnO  = attnO;

  // 1) weight transpose + f16 convert
  wtrans_kernel<<<dim3(DMODEL / 16, DMODEL / 16), dim3(16, 16), 0, stream>>>(
      Wk, WkT, DMODEL, DMODEL);
  wtrans_kernel<<<dim3(DMODEL / 16, DMODEL / 16), dim3(16, 16), 0, stream>>>(
      Wv, WvT, DMODEL, DMODEL);
  wtrans_kernel<<<dim3(DMODEL / 16, DMODEL / 16), dim3(16, 16), 0, stream>>>(
      Wo, WoT, DMODEL, DMODEL);
  wtrans_kernel<<<dim3(DFF / 16, DMODEL / 16), dim3(16, 16), 0, stream>>>(
      W1, W1T, DMODEL, DFF);
  wtrans_kernel<<<dim3(DMODEL / 16, DFF / 16), dim3(16, 16), 0, stream>>>(
      W2, W2T, DFF, DMODEL);

  // 2) activations -> f16
  const int nAct = ROWS * DMODEL;
  cvt_f32_to_f16<<<(nAct + 255) / 256, 256, 0, stream>>>(query, qry16, nAct);
  cvt_f32_to_f16<<<(nAct + 255) / 256, 256, 0, stream>>>(key, key16, nAct);
  cvt_f32_to_f16<<<(nAct + 255) / 256, 256, 0, stream>>>(values, val16, nAct);

  // 3) QKV projections (pipelined block GEMM, head-split epilogues)
  gemm_wmma<<<dim3(DMODEL / 64, ROWS / 64), 128, 0, stream>>>(
      qry16, WkT, bk, (float*)nullptr, q16, DMODEL, DMODEL, 0, 1);
  gemm_wmma<<<dim3(DMODEL / 64, ROWS / 64), 128, 0, stream>>>(
      key16, WkT, bk, (float*)nullptr, k16, DMODEL, DMODEL, 0, 1);
  gemm_wmma<<<dim3(DMODEL / 64, ROWS / 64), 128, 0, stream>>>(
      val16, WvT, bv, (float*)nullptr, vT16, DMODEL, DMODEL, 0, 2);

  // 4) attention -> ctx16
  attn_kernel<<<dim3(SEQ / 16, BATCH * NHEADS), 32, 0, stream>>>(
      q16, k16, vT16, dif, a_p, al_p, maskp, ctx16);

  // 5) output projection Wo
  gemm_wmma<<<dim3(DMODEL / 64, ROWS / 64), 128, 0, stream>>>(
      ctx16, WoT, bo, attnO, (_Float16*)nullptr, DMODEL, DMODEL, 0, 0);

  // 6) LN1(query + attnO) -> xbuf (f32) + x16
  ln_kernel<<<ROWS, 256, 0, stream>>>(query, attnO, ln1g, ln1b, xbuf, x16);

  // 7) FFN: relu(x @ W1 + b1) -> h16 ; h @ W2 + b2 -> ffnO
  gemm_wmma<<<dim3(DFF / 64, ROWS / 64), 128, 0, stream>>>(
      x16, W1T, b1, (float*)nullptr, h16, DFF, DMODEL, 1, 0);
  gemm_wmma<<<dim3(DMODEL / 64, ROWS / 64), 128, 0, stream>>>(
      h16, W2T, b2, ffnO, (_Float16*)nullptr, DMODEL, DFF, 0, 0);

  // 8) LN2(x + ffn) -> d_out
  ln_kernel<<<ROWS, 256, 0, stream>>>(xbuf, ffnO, ln2g, ln2b, (float*)d_out,
                                      (_Float16*)nullptr);
}
